// CrossTransformer_16836271801134
// MI455X (gfx1250) — matrix operand order repaired and checked by reading, not compiled, not run on hardware
//
#include <hip/hip_runtime.h>
#include <hip/hip_bf16.h>

typedef __attribute__((ext_vector_type(16))) _Float16 v16h;
typedef __attribute__((ext_vector_type(8)))  _Float16 v8h;
typedef __attribute__((ext_vector_type(8)))  float    v8f;

#define BN_EPS 1e-5f

constexpr int Bb = 2, Cc = 256, Nn = 2048, Ff = 4096, Kk = 16, PH = 64, AH = 1024;
constexpr int Tt = Bb * Nn * Kk;          // 65536 tokens
constexpr int NSPLIT = 8;
constexpr int CAND = Ff / NSPLIT;         // 512 candidates per split

// ---------------- workspace layout (bytes) ----------------
constexpr size_t OFF_IDX = 0;
constexpr size_t SZ_IDX  = (size_t)Bb * Nn * Kk * 4;          // 256 KB
constexpr size_t OFF_PD  = OFF_IDX + SZ_IDX;
constexpr size_t SZ_PD   = (size_t)Bb * Nn * NSPLIT * 16 * 4; // 2 MB
constexpr size_t OFF_PI  = OFF_PD + SZ_PD;
constexpr size_t OFF_W1H = OFF_PI + SZ_PD;
constexpr size_t SZ_WH   = (size_t)AH * Cc * 2;               // 512 KB each
constexpr size_t OFF_W2H = OFF_W1H + SZ_WH;
constexpr size_t OFF_B1F = OFF_W2H + SZ_WH;
constexpr size_t OFF_PW1 = OFF_B1F + 4096;
constexpr size_t OFF_PB1 = OFF_PW1 + 1024;
constexpr size_t OFF_FT  = OFF_PB1 + 256;
constexpr size_t SZ_FT   = (size_t)Bb * Ff * Cc * 4;          // 8 MB
constexpr size_t OFF_X   = OFF_FT + SZ_FT;
constexpr size_t SZ_X    = (size_t)Tt * Cc * 2;               // 32 MB
constexpr size_t OFF_V   = OFF_X + SZ_X;
constexpr size_t OFF_S   = OFF_V + SZ_X;                      // S: 64 MB f32

// ============================================================
// K0: fold BN into weights, pack W1/W2 into WMMA B-fragment layout (f16)
// B fragment (K=32 x N=16 f16): lane L holds column n=L%16, K = (j/8)*16 + (L/16)*8 + j%8
// packed as frag-major, lane-major, 16 f16 contiguous per lane.
// ============================================================
__global__ void ct_prep_kernel(const float* __restrict__ aw1, const float* __restrict__ ab1,
                               const float* __restrict__ ag1, const float* __restrict__ abe1,
                               const float* __restrict__ aw2,
                               const float* __restrict__ pw1, const float* __restrict__ pb1,
                               const float* __restrict__ pg1, const float* __restrict__ pbe1,
                               _Float16* __restrict__ W1h, _Float16* __restrict__ W2h,
                               float* __restrict__ b1f, float* __restrict__ pW1f,
                               float* __restrict__ pb1f) {
  int e = blockIdx.x * 256 + threadIdx.x;
  const float inv = rsqrtf(1.f + BN_EPS);
  if (e < AH * Cc) { // 262144: W1 fragments [64 htiles][8 kchunks][32 lanes][16]
    int j = e & 15, L = (e >> 4) & 31, kc = (e >> 9) & 7, ht = e >> 12;
    int row = ht * 16 + (L & 15);             // hidden unit
    int col = kc * 32 + (j >> 3) * 16 + (L >> 4) * 8 + (j & 7);    // input channel (K)
    float s = ag1[row] * inv;
    W1h[e] = (_Float16)(aw1[row * Cc + col] * s);
    return;
  }
  e -= AH * Cc;
  if (e < Cc * AH) { // 262144: W2 fragments [16 cotiles][32 hchunks][32 lanes][16]
    int j = e & 15, L = (e >> 4) & 31, hc = (e >> 9) & 31, ct = e >> 14;
    int row = ct * 16 + (L & 15);             // output channel
    int col = hc * 32 + (j >> 3) * 16 + (L >> 4) * 8 + (j & 7);    // hidden (K)
    W2h[e] = (_Float16)(aw2[row * AH + col]);
    return;
  }
  e -= Cc * AH;
  if (e < AH) { float s = ag1[e] * inv; b1f[e] = ab1[e] * s + abe1[e]; return; }
  e -= AH;
  if (e < PH * 3) { int h = e / 3; pW1f[e] = pw1[e] * (pg1[h] * inv); return; }
  e -= PH * 3;
  if (e < PH) { pb1f[e] = pb1[e] * (pg1[e] * inv) + pbe1[e]; return; }
}

// ============================================================
// K1: transpose fusion_feat -> featT [b][F][C] for coalesced gathers
// ============================================================
__global__ void ct_transpose_kernel(const float* __restrict__ feat,
                                    const float* __restrict__ featdb,
                                    float* __restrict__ featT) {
  int e = blockIdx.x * 256 + threadIdx.x;       // b*F*C = 2^21
  int c = e & 255, f = (e >> 8) & (Ff - 1), b = e >> 20;
  float v = (f < Nn) ? feat[((size_t)b * Cc + c) * Nn + f]
                     : featdb[((size_t)b * Cc + c) * Nn + (f - Nn)];
  featT[e] = v;
}

// ---------------- register-resident sorted top-16 insert (static indices) ----
__device__ __forceinline__ void topk_insert(float (&bd)[16], int (&bi)[16], float d, int f) {
  if (d < bd[15]) {
    bd[15] = d; bi[15] = f;
#pragma unroll
    for (int i = 15; i > 0; --i) {
      if (bd[i] < bd[i - 1]) {
        float td = bd[i]; bd[i] = bd[i - 1]; bd[i - 1] = td;
        int   ti = bi[i]; bi[i] = bi[i - 1]; bi[i - 1] = ti;
      }
    }
  }
}

// ============================================================
// K2a: partial KNN, 8-way split of the candidate range (parallelism)
// ============================================================
__global__ void ct_knn_part_kernel(const float* __restrict__ pcd,
                                   const float* __restrict__ pcddb,
                                   float* __restrict__ pd, int* __restrict__ pi) {
  __shared__ float lx[CAND], ly[CAND], lz[CAND];
  int blk = blockIdx.x, sp = blk & (NSPLIT - 1), qg = blk >> 3;
  int q = qg * 256 + threadIdx.x;               // global query 0..4095
  int b = q >> 11, n = q & (Nn - 1);
  for (int i = threadIdx.x; i < CAND; i += 256) {
    int f = sp * CAND + i;
    const float* src = (f < Nn) ? pcd : pcddb;
    int fl = (f < Nn) ? f : f - Nn;
    lx[i] = src[((size_t)b * 3 + 0) * Nn + fl];
    ly[i] = src[((size_t)b * 3 + 1) * Nn + fl];
    lz[i] = src[((size_t)b * 3 + 2) * Nn + fl];
  }
  __syncthreads();
  float qx = pcd[((size_t)b * 3 + 0) * Nn + n];
  float qy = pcd[((size_t)b * 3 + 1) * Nn + n];
  float qz = pcd[((size_t)b * 3 + 2) * Nn + n];
  float bd[16]; int bi[16];
#pragma unroll
  for (int i = 0; i < 16; ++i) { bd[i] = 3.4e38f; bi[i] = 0; }
  for (int i = 0; i < CAND; ++i) {
    float dx = qx - lx[i], dy = qy - ly[i], dz = qz - lz[i];
    topk_insert(bd, bi, dx * dx + dy * dy + dz * dz, sp * CAND + i);
  }
#pragma unroll
  for (int i = 0; i < 16; ++i) {
    pd[((size_t)q * NSPLIT + sp) * 16 + i] = bd[i];
    pi[((size_t)q * NSPLIT + sp) * 16 + i] = bi[i];
  }
}

// ============================================================
// K2b: merge the 8 partial top-16 lists per query
// ============================================================
__global__ void ct_knn_merge_kernel(const float* __restrict__ pd,
                                    const int* __restrict__ pi,
                                    int* __restrict__ idx) {
  int q = blockIdx.x * 256 + threadIdx.x;
  float bd[16]; int bi[16];
#pragma unroll
  for (int i = 0; i < 16; ++i) { bd[i] = 3.4e38f; bi[i] = 0; }
  for (int s = 0; s < NSPLIT; ++s)
    for (int i = 0; i < 16; ++i)
      topk_insert(bd, bi, pd[((size_t)q * NSPLIT + s) * 16 + i],
                          pi[((size_t)q * NSPLIT + s) * 16 + i]);
#pragma unroll
  for (int i = 0; i < 16; ++i) idx[(size_t)q * 16 + i] = bi[i];
}

// ============================================================
// K3: gather + pos-MLP (3->64->256), emit X = qk_rel + pos_emb (f16)
//     and V = group_feat + pos_emb (f16). One block per (b,n).
// ============================================================
__global__ void ct_build_tokens_kernel(const float* __restrict__ pcd,
                                       const float* __restrict__ pcddb,
                                       const float* __restrict__ feat,
                                       const float* __restrict__ featT,
                                       const int* __restrict__ idx,
                                       const float* __restrict__ pW1f,
                                       const float* __restrict__ pb1f,
                                       const float* __restrict__ pw2,
                                       const float* __restrict__ pb2,
                                       _Float16* __restrict__ X,
                                       _Float16* __restrict__ V) {
  __shared__ float ls_h[Kk][PH];
  __shared__ float ls_prel[Kk][3];
  __shared__ int   ls_idx[Kk];
  __shared__ float ls_featn[Cc];
  int bn = blockIdx.x, b = bn >> 11, n = bn & (Nn - 1);
  int tid = threadIdx.x;
  ls_featn[tid] = feat[((size_t)b * Cc + tid) * Nn + n];
  if (tid < Kk) ls_idx[tid] = idx[(size_t)bn * Kk + tid];
  __syncthreads();
  if (tid < Kk * 3) {
    int k = tid / 3, d = tid - k * 3;
    int f = ls_idx[k];
    float pv = (f < Nn) ? pcd[((size_t)b * 3 + d) * Nn + f]
                        : pcddb[((size_t)b * 3 + d) * Nn + (f - Nn)];
    ls_prel[k][d] = pcd[((size_t)b * 3 + d) * Nn + n] - pv;
  }
  __syncthreads();
#pragma unroll
  for (int r = 0; r < 4; ++r) {                 // 16*64 = 1024 hidden values
    int id = tid + 256 * r;
    int k = id >> 6, j = id & 63;
    float h = pb1f[j] + pW1f[j * 3 + 0] * ls_prel[k][0]
                      + pW1f[j * 3 + 1] * ls_prel[k][1]
                      + pW1f[j * 3 + 2] * ls_prel[k][2];
    ls_h[k][j] = fmaxf(h, 0.f);
  }
  __syncthreads();
  int c = tid;
  float pe[Kk];
#pragma unroll
  for (int k = 0; k < Kk; ++k) pe[k] = pb2[c];
  for (int j = 0; j < PH; ++j) {
    float w = pw2[c * PH + j];
#pragma unroll
    for (int k = 0; k < Kk; ++k) pe[k] += w * ls_h[k][j];
  }
  float fc = ls_featn[c];
#pragma unroll
  for (int k = 0; k < Kk; ++k) {
    int f = ls_idx[k];
    float gf = featT[((size_t)b * Ff + f) * Cc + c];
    size_t t = (size_t)bn * Kk + k;
    X[t * Cc + c] = (_Float16)(fc - gf + pe[k]);
    V[t * Cc + c] = (_Float16)(gf + pe[k]);
  }
}

// ============================================================
// K4: fused attention MLP  S = (relu(bn(X@W1^T)) @ W2^T + b2)
//     512 blocks x 8 waves; wave owns 16 token rows x 256 out cols.
//     Hidden (1024) processed in chunks of 32, never touches HBM.
//     X tile staged with CDNA5 async global->LDS DMA (ASYNCcnt).
// ============================================================
__global__ __launch_bounds__(256) void ct_attn_mlp_kernel(
    const _Float16* __restrict__ X, const _Float16* __restrict__ W1h,
    const _Float16* __restrict__ W2h, const float* __restrict__ b1f,
    const float* __restrict__ b2, float* __restrict__ S) {
  constexpr int PITCH  = 264;  // f16 units; 528B row stride -> conflict-free b128
  constexpr int HPITCH = 40;   // f16 units
  __shared__ _Float16 lsX[128 * PITCH];        // 67.5 KB X tile
  __shared__ _Float16 lsH[8 * 16 * HPITCH];    // 10 KB per-wave H scratch
  int tid = threadIdx.x;
  int wave = tid >> 5, lane = tid & 31;
  int lrow = lane & 15, lhi = lane >> 4;
  int t0 = blockIdx.x * 128;

  // stage X tile: 128 rows x 256 f16 via async DMA straight into LDS
  // (global_load_async_to_lds_b128: per-lane 16B memory->LDS, ASYNCcnt)
  {
    unsigned ldsbase = (unsigned)(size_t)(&lsX[0]);
#pragma unroll 4
    for (int i = tid; i < 128 * 32; i += 256) {
      int row = i >> 5, cb = i & 31;
      const _Float16* gp = X + (size_t)(t0 + row) * Cc + cb * 8;
      unsigned la = ldsbase + (unsigned)(row * PITCH + cb * 8) * 2u;
      asm volatile("global_load_async_to_lds_b128 %0, %1, off"
                   :: "v"(la), "v"(gp) : "memory");
    }
    asm volatile("s_wait_asynccnt 0x0" ::: "memory");
  }
  __syncthreads();

  const _Float16* lsXrow = lsX + (wave * 16 + lrow) * PITCH;
  _Float16* lsHw = lsH + wave * 16 * HPITCH;   // wave-private

  v8f acc[16] = {};                            // 16 co-tiles x 8 VGPR f32

  for (int hc = 0; hc < 32; ++hc) {            // hidden chunk of 32
    // ---- GEMM1: H[16x32] = Xtile[16x256] @ W1^T[256x32] ----
    v8f hacc0 = {}, hacc1 = {};
#pragma unroll
    for (int kc = 0; kc < 8; ++kc) {
      union { v16h v; v8h h[2]; } a;           // A frag 16x32 f16
      a.h[0] = *(const v8h*)(lsXrow + kc * 32 + lhi * 8);
      a.h[1] = *(const v8h*)(lsXrow + kc * 32 + 16 + lhi * 8);
      const _Float16* bp0 = W1h + ((size_t)((hc * 2 + 0) * 8 + kc) * 512) + lane * 16;
      const _Float16* bp1 = W1h + ((size_t)((hc * 2 + 1) * 8 + kc) * 512) + lane * 16;
      union { v16h v; v8h h[2]; } b0, b1;
      b0.h[0] = *(const v8h*)(bp0); b0.h[1] = *(const v8h*)(bp0 + 8);
      b1.h[0] = *(const v8h*)(bp1); b1.h[1] = *(const v8h*)(bp1 + 8);
      hacc0 = __builtin_amdgcn_wmma_f32_16x16x32_f16(false, a.v, false, b0.v,
                                                     (short)0, hacc0, false, false);
      hacc1 = __builtin_amdgcn_wmma_f32_16x16x32_f16(false, a.v, false, b1.v,
                                                     (short)0, hacc1, false, false);
    }
    // bias + relu in D layout; re-fragment D->A via wave-private LDS
    {
      float bia0 = b1f[hc * 32 + lrow];
      float bia1 = b1f[hc * 32 + 16 + lrow];
#pragma unroll
      for (int v = 0; v < 8; ++v) {
        int m = v + lhi * 8;
        lsHw[m * HPITCH + lrow]      = (_Float16)fmaxf(hacc0[v] + bia0, 0.f);
        lsHw[m * HPITCH + 16 + lrow] = (_Float16)fmaxf(hacc1[v] + bia1, 0.f);
      }
    }
    asm volatile("s_wait_dscnt 0" ::: "memory");   // wave-local LDS RAW
    union { v16h v; v8h h[2]; } a2;                // A frag 16x32 of H
    a2.h[0] = *(const v8h*)(lsHw + lrow * HPITCH + lhi * 8);
    a2.h[1] = *(const v8h*)(lsHw + lrow * HPITCH + 16 + lhi * 8);
    // ---- GEMM2: acc[16 tiles] += H[16x32] @ W2^T[32x256] ----
#pragma unroll
    for (int ct = 0; ct < 16; ++ct) {
      const _Float16* bp = W2h + ((size_t)(ct * 32 + hc) * 512) + lane * 16;
      union { v16h v; v8h h[2]; } bb;
      bb.h[0] = *(const v8h*)(bp); bb.h[1] = *(const v8h*)(bp + 8);
      acc[ct] = __builtin_amdgcn_wmma_f32_16x16x32_f16(false, a2.v, false, bb.v,
                                                       (short)0, acc[ct], false, false);
    }
  }
  // epilogue: + b2, store attention logits S [T x 256] f32
#pragma unroll
  for (int ct = 0; ct < 16; ++ct) {
    float bia = b2[ct * 16 + lrow];
#pragma unroll
    for (int v = 0; v < 8; ++v) {
      int m = v + lhi * 8;
      int t = t0 + wave * 16 + m;
      S[(size_t)t * Cc + ct * 16 + lrow] = acc[ct][v] + bia;
    }
  }
}

// ============================================================
// K5: softmax over k + weighted sum -> out [b, C, N]
// ============================================================
__global__ void ct_finalize_kernel(const float* __restrict__ S,
                                   const _Float16* __restrict__ V,
                                   float* __restrict__ out) {
  int bn = blockIdx.x, c = threadIdx.x;
  int b = bn >> 11, n = bn & (Nn - 1);
  size_t base = (size_t)bn * Kk;
  float sv[Kk], mx = -3.4e38f;
#pragma unroll
  for (int k = 0; k < Kk; ++k) { sv[k] = S[(base + k) * Cc + c]; mx = fmaxf(mx, sv[k]); }
  float sum = 0.f;
#pragma unroll
  for (int k = 0; k < Kk; ++k) { sv[k] = __expf(sv[k] - mx); sum += sv[k]; }
  float inv = 1.f / sum, r = 0.f;
#pragma unroll
  for (int k = 0; k < Kk; ++k) r += sv[k] * inv * (float)V[(base + k) * Cc + c];
  out[((size_t)b * Cc + c) * Nn + n] = r;
}

// ============================================================
extern "C" void kernel_launch(void* const* d_in, const int* in_sizes, int n_in,
                              void* d_out, int out_size, void* d_ws, size_t ws_size,
                              hipStream_t stream) {
  const float* pcd      = (const float*)d_in[0];
  const float* feat     = (const float*)d_in[1];
  const float* pcd_db   = (const float*)d_in[2];
  const float* feat_db  = (const float*)d_in[3];
  const float* pos_w1   = (const float*)d_in[4];
  const float* pos_b1   = (const float*)d_in[5];
  const float* pos_g1   = (const float*)d_in[6];
  const float* pos_be1  = (const float*)d_in[7];
  const float* pos_w2   = (const float*)d_in[8];
  const float* pos_b2   = (const float*)d_in[9];
  const float* attn_w1  = (const float*)d_in[10];
  const float* attn_b1  = (const float*)d_in[11];
  const float* attn_g1  = (const float*)d_in[12];
  const float* attn_be1 = (const float*)d_in[13];
  const float* attn_w2  = (const float*)d_in[14];
  const float* attn_b2  = (const float*)d_in[15];

  char* ws = (char*)d_ws;
  int*      idx  = (int*)(ws + OFF_IDX);
  float*    pd   = (float*)(ws + OFF_PD);
  int*      pi   = (int*)(ws + OFF_PI);
  _Float16* W1h  = (_Float16*)(ws + OFF_W1H);
  _Float16* W2h  = (_Float16*)(ws + OFF_W2H);
  float*    b1f  = (float*)(ws + OFF_B1F);
  float*    pW1f = (float*)(ws + OFF_PW1);
  float*    pb1f = (float*)(ws + OFF_PB1);
  float*    ftT  = (float*)(ws + OFF_FT);
  _Float16* X    = (_Float16*)(ws + OFF_X);
  _Float16* V    = (_Float16*)(ws + OFF_V);
  float*    S    = (float*)(ws + OFF_S);
  float*    out  = (float*)d_out;

  int prep_total = AH * Cc + Cc * AH + AH + PH * 3 + PH;
  ct_prep_kernel<<<(prep_total + 255) / 256, 256, 0, stream>>>(
      attn_w1, attn_b1, attn_g1, attn_be1, attn_w2,
      pos_w1, pos_b1, pos_g1, pos_be1, W1h, W2h, b1f, pW1f, pb1f);
  ct_transpose_kernel<<<(Bb * Ff * Cc) / 256, 256, 0, stream>>>(feat, feat_db, ftT);
  ct_knn_part_kernel<<<(Bb * Nn / 256) * NSPLIT, 256, 0, stream>>>(pcd, pcd_db, pd, pi);
  ct_knn_merge_kernel<<<(Bb * Nn) / 256, 256, 0, stream>>>(pd, pi, idx);
  ct_build_tokens_kernel<<<Bb * Nn, 256, 0, stream>>>(
      pcd, pcd_db, feat, ftT, idx, pW1f, pb1f, pos_w2, pos_b2, X, V);
  ct_attn_mlp_kernel<<<Tt / 128, 256, 0, stream>>>(X, W1h, W2h, b1f, attn_b2, S);
  ct_finalize_kernel<<<Bb * Nn, 256, 0, stream>>>(S, V, out);
}
